// SimpleMambaBlock_29729763623374
// MI455X (gfx1250) — compile-verified, matrix operand order checked by reading, not run on hardware
//
#include <hip/hip_runtime.h>
#include <hip/hip_bf16.h>

// ---------------------------------------------------------------------------
// SimpleMambaBlock for MI455X (gfx1250, wave32, WMMA f32_16x16x32_f16)
// 2x4 register-blocked WMMA GEMMs: each wave computes a 32x64 output tile.
// ---------------------------------------------------------------------------

#define BATCH 8
#define SEQL  2048
#define DM    512
#define DI    1024
#define DS    16
#define MROWS (BATCH * SEQL)   // 16384
#define MT2   (MROWS / 32)     // 512 (32-row M tiles)

typedef _Float16 v8h  __attribute__((ext_vector_type(8)));
typedef _Float16 v16h __attribute__((ext_vector_type(16)));
typedef float    v8f  __attribute__((ext_vector_type(8)));

__device__ __forceinline__ float sigmoidf_(float x) { return 1.0f / (1.0f + __expf(-x)); }
__device__ __forceinline__ float siluf_(float x)    { return x * sigmoidf_(x); }

// ---- WMMA fragment loader -------------------------------------------------
// 16-bit A (16x32) / B (32x16 as rows of W) fragment, per CDNA5 ISA 7.12.2:
//   lanes 0-15  : row = lane,      halves 0-7 = K 0..7,  halves 8-15 = K 16..23
//   lanes 16-31 : row = lane-16,   halves 0-7 = K 8..15, halves 8-15 = K 24..31
__device__ __forceinline__ v16h load_frag16(const _Float16* __restrict__ rowp, int lane) {
  const int koff = (lane & 16) ? 8 : 0;
  v8h lo = *(const v8h*)(rowp + koff);
  v8h hi = *(const v8h*)(rowp + 16 + koff);
  v16h r;
#pragma unroll
  for (int i = 0; i < 8; ++i) { r[i] = lo[i]; r[8 + i] = hi[i]; }
  return r;
}

// Wave-level 32x(16*TN) tile: D = A[m0:+32, :K] * B[n0:+16*TN, :K]^T
// 2 A-frags + TN B-frags per K-step feed 2*TN WMMAs.
template <int TN>
__device__ __forceinline__ void wmma_tile2(const _Float16* __restrict__ A,
                                           const _Float16* __restrict__ B,
                                           int lda, int ldb, int K,
                                           int m0, int n0, int lane, v8f acc[2][TN]) {
  const _Float16* arow0 = A + (size_t)(m0 + (lane & 15)) * lda;
  const _Float16* arow1 = arow0 + (size_t)16 * lda;
  const _Float16* brow  = B + (size_t)(n0 + (lane & 15)) * ldb;
  for (int k = 0; k < K; k += 32) {
    if (k + 32 < K) {
      __builtin_prefetch(arow0 + k + 32, 0, 0);    // global_prefetch_b8
      __builtin_prefetch(arow1 + k + 32, 0, 0);
      __builtin_prefetch(brow + k + 32, 0, 0);
    }
    v16h a0 = load_frag16(arow0 + k, lane);
    v16h a1 = load_frag16(arow1 + k, lane);
#pragma unroll
    for (int t = 0; t < TN; ++t) {
      v16h b = load_frag16(brow + (size_t)(16 * t) * ldb + k, lane);
      acc[0][t] = __builtin_amdgcn_wmma_f32_16x16x32_f16(
          false, a0, false, b, (short)0, acc[0][t], false, false);
      acc[1][t] = __builtin_amdgcn_wmma_f32_16x16x32_f16(
          false, a1, false, b, (short)0, acc[1][t], false, false);
    }
  }
}

// C/D mapping (16x16 f32 tile): elem v of lane l -> (v + 8*((l>>4)&1), l&15)

// ---- misc elementwise kernels ---------------------------------------------

__global__ void k_f32_to_f16(const float* __restrict__ s, _Float16* __restrict__ d, int n) {
  int i = blockIdx.x * blockDim.x + threadIdx.x;
  if (i < n) d[i] = (_Float16)s[i];
}

// ssm_proj_w [DI, 16] -> f16 [DI, 32] zero K-padded
__global__ void k_ssm_pad(const float* __restrict__ s, _Float16* __restrict__ d) {
  int i = blockIdx.x * blockDim.x + threadIdx.x;  // over DI*32
  if (i >= DI * 32) return;
  int e = i >> 5, c = i & 31;
  d[i] = (c < DS) ? (_Float16)s[e * DS + c] : (_Float16)0.0f;
}

// ---- LayerNorm -> f16 -----------------------------------------------------
__global__ void k_layernorm(const float* __restrict__ x, const float* __restrict__ w,
                            const float* __restrict__ b, _Float16* __restrict__ out) {
  __shared__ float s1[256], s2[256];
  const int row = blockIdx.x, t = threadIdx.x;
  const float* xr = x + (size_t)row * DM;
  float a0 = xr[t], a1 = xr[t + 256];
  s1[t] = a0 + a1;
  s2[t] = a0 * a0 + a1 * a1;
  __syncthreads();
  for (int s = 128; s > 0; s >>= 1) {
    if (t < s) { s1[t] += s1[t + s]; s2[t] += s2[t + s]; }
    __syncthreads();
  }
  float mu  = s1[0] * (1.0f / DM);
  float var = s2[0] * (1.0f / DM) - mu * mu;
  float inv = rsqrtf(var + 1e-5f);
  _Float16* orow = out + (size_t)row * DM;
  orow[t]       = (_Float16)((a0 - mu) * inv * w[t] + b[t]);
  orow[t + 256] = (_Float16)((a1 - mu) * inv * w[t + 256] + b[t + 256]);
}

// ---- GEMM1: xz = xn @ in_proj_w^T (+bias), split -> x_ f16, z f16 ----------
__global__ void k_gemm_inproj(const _Float16* __restrict__ A, const _Float16* __restrict__ W,
                              const float* __restrict__ bias,
                              _Float16* __restrict__ x16, _Float16* __restrict__ z16) {
  const int lane = threadIdx.x & 31;
  const int wid  = blockIdx.x * (blockDim.x >> 5) + (threadIdx.x >> 5);
  const int mt = wid % MT2, ng = wid / MT2;        // 32 n-groups of 64
  const int m0 = mt * 32, n0 = ng * 64;
  v8f acc[2][4] = {};
  wmma_tile2<4>(A, W, DM, DM, DM, m0, n0, lane, acc);
  const int nl  = lane & 15;
  const int moff = ((lane >> 4) & 1) * 8;
#pragma unroll
  for (int mi = 0; mi < 2; ++mi) {
    const int mlo = m0 + mi * 16 + moff;
#pragma unroll
    for (int t = 0; t < 4; ++t) {
      int n = n0 + 16 * t + nl;
      float bi = bias[n];
#pragma unroll
      for (int v = 0; v < 8; ++v) {
        int m = mlo + v;
        float val = acc[mi][t][v] + bi;
        if (n < DI) x16[(size_t)m * DI + n]        = (_Float16)val;
        else        z16[(size_t)m * DI + (n - DI)] = (_Float16)val;
      }
    }
  }
}

// ---- depthwise conv1d (k=3, pad=1) + SiLU ---------------------------------
__global__ void k_conv_silu(const _Float16* __restrict__ xi, const float* __restrict__ cw,
                            const float* __restrict__ cb, _Float16* __restrict__ xo) {
  int i = blockIdx.x * blockDim.x + threadIdx.x;   // over MROWS*DI
  if (i >= MROWS * DI) return;
  int c = i & (DI - 1);
  int ml = i >> 10;                                 // b*SEQL + l
  int l = ml & (SEQL - 1);
  float acc = cb[c];
  if (l > 0)        acc += (float)xi[(size_t)(ml - 1) * DI + c] * cw[c * 3 + 0];
  acc              += (float)xi[(size_t)ml * DI + c]            * cw[c * 3 + 1];
  if (l < SEQL - 1) acc += (float)xi[(size_t)(ml + 1) * DI + c] * cw[c * 3 + 2];
  xo[i] = (_Float16)siluf_(acc);
}

// ---- GEMM2: xg = xconv @ gru_w_ih^T (+b_ih) -> fp32 [MROWS,48] ------------
__global__ void k_gemm_gru(const _Float16* __restrict__ A, const _Float16* __restrict__ W,
                           const float* __restrict__ bias, float* __restrict__ xg) {
  const int lane = threadIdx.x & 31;
  const int wid  = blockIdx.x * (blockDim.x >> 5) + (threadIdx.x >> 5);
  const int m0 = (wid % MT2) * 32;
  v8f acc[2][3] = {};
  wmma_tile2<3>(A, W, DI, DI, DI, m0, 0, lane, acc);
  const int nl  = lane & 15;
  const int moff = ((lane >> 4) & 1) * 8;
#pragma unroll
  for (int mi = 0; mi < 2; ++mi) {
    const int mlo = m0 + mi * 16 + moff;
#pragma unroll
    for (int t = 0; t < 3; ++t) {
      int n = 16 * t + nl;
      float bi = bias[n];
#pragma unroll
      for (int v = 0; v < 8; ++v)
        xg[(size_t)(mlo + v) * 48 + n] = acc[mi][t][v] + bi;
    }
  }
}

// ---- GRU scan (serial over L), writes K-padded f16 y [MROWS, 32] ----------
__global__ void k_gru_scan(const float* __restrict__ xg, const float* __restrict__ whh,
                           const float* __restrict__ bhh, _Float16* __restrict__ ypad) {
  __shared__ float hsh[BATCH][DS];
  __shared__ float sw[48 * DS];
  __shared__ float sb[48];
  const int t = threadIdx.x;                  // 128 threads
  for (int i = t; i < 48 * DS; i += 128) sw[i] = whh[i];
  if (t < 48) sb[t] = bhh[t];
  const int b = t >> 4, h = t & 15;
  hsh[b][h] = 0.0f;
  __syncthreads();
  for (int l = 0; l < SEQL; ++l) {
    float hr = sb[h], hz = sb[16 + h], hn = sb[32 + h];
#pragma unroll
    for (int j = 0; j < DS; ++j) {
      float hj = hsh[b][j];
      hr += sw[h * DS + j] * hj;
      hz += sw[(16 + h) * DS + j] * hj;
      hn += sw[(32 + h) * DS + j] * hj;
    }
    const float* xr = xg + (size_t)(b * SEQL + l) * 48;
    float r  = sigmoidf_(xr[h] + hr);
    float zt = sigmoidf_(xr[16 + h] + hz);
    float nn = tanhf(xr[32 + h] + r * hn);
    float hnew = (1.0f - zt) * nn + zt * hsh[b][h];
    __syncthreads();
    hsh[b][h] = hnew;
    __syncthreads();
    _Float16* yr = ypad + (size_t)(b * SEQL + l) * 32;
    yr[h]      = (_Float16)hnew;
    yr[16 + h] = (_Float16)0.0f;
  }
}

// ---- GEMM3: (y @ ssm_w^T + b) * silu(z) -> gated f16 [MROWS, DI] ----------
__global__ void k_gemm_ssm_gate(const _Float16* __restrict__ A, const _Float16* __restrict__ W,
                                const float* __restrict__ bias, const _Float16* __restrict__ z16,
                                _Float16* __restrict__ gated) {
  const int lane = threadIdx.x & 31;
  const int wid  = blockIdx.x * (blockDim.x >> 5) + (threadIdx.x >> 5);
  const int mt = wid % MT2, ng = wid / MT2;   // 16 n-groups of 64
  const int m0 = mt * 32, n0 = ng * 64;
  v8f acc[2][4] = {};
  wmma_tile2<4>(A, W, 32, 32, 32, m0, n0, lane, acc);
  const int nl  = lane & 15;
  const int moff = ((lane >> 4) & 1) * 8;
#pragma unroll
  for (int mi = 0; mi < 2; ++mi) {
    const int mlo = m0 + mi * 16 + moff;
#pragma unroll
    for (int t = 0; t < 4; ++t) {
      int n = n0 + 16 * t + nl;
      float bi = bias[n];
#pragma unroll
      for (int v = 0; v < 8; ++v) {
        int m = mlo + v;
        float zf = (float)z16[(size_t)m * DI + n];
        gated[(size_t)m * DI + n] = (_Float16)((acc[mi][t][v] + bi) * siluf_(zf));
      }
    }
  }
}

// ---- GEMM4: out = gated @ out_proj_w^T + b + residual -> fp32 -------------
__global__ void k_gemm_out(const _Float16* __restrict__ A, const _Float16* __restrict__ W,
                           const float* __restrict__ bias, const float* __restrict__ resid,
                           float* __restrict__ out) {
  const int lane = threadIdx.x & 31;
  const int wid  = blockIdx.x * (blockDim.x >> 5) + (threadIdx.x >> 5);
  const int mt = wid % MT2, ng = wid / MT2;   // 8 n-groups of 64
  const int m0 = mt * 32, n0 = ng * 64;
  v8f acc[2][4] = {};
  wmma_tile2<4>(A, W, DI, DI, DI, m0, n0, lane, acc);
  const int nl  = lane & 15;
  const int moff = ((lane >> 4) & 1) * 8;
#pragma unroll
  for (int mi = 0; mi < 2; ++mi) {
    const int mlo = m0 + mi * 16 + moff;
#pragma unroll
    for (int t = 0; t < 4; ++t) {
      int n = n0 + 16 * t + nl;
      float bi = bias[n];
#pragma unroll
      for (int v = 0; v < 8; ++v) {
        size_t idx = (size_t)(mlo + v) * DM + n;
        out[idx] = acc[mi][t][v] + bi + resid[idx];
      }
    }
  }
}

// ---------------------------------------------------------------------------
extern "C" void kernel_launch(void* const* d_in, const int* in_sizes, int n_in,
                              void* d_out, int out_size, void* d_ws, size_t ws_size,
                              hipStream_t stream) {
  (void)in_sizes; (void)n_in; (void)out_size; (void)ws_size;
  const float* x         = (const float*)d_in[0];
  const float* norm_w    = (const float*)d_in[1];
  const float* norm_b    = (const float*)d_in[2];
  const float* in_proj_w = (const float*)d_in[3];
  const float* in_proj_b = (const float*)d_in[4];
  const float* conv_w    = (const float*)d_in[5];
  const float* conv_b    = (const float*)d_in[6];
  const float* gru_w_ih  = (const float*)d_in[7];
  const float* gru_w_hh  = (const float*)d_in[8];
  const float* gru_b_ih  = (const float*)d_in[9];
  const float* gru_b_hh  = (const float*)d_in[10];
  const float* ssm_w     = (const float*)d_in[11];
  const float* ssm_b     = (const float*)d_in[12];
  const float* out_w     = (const float*)d_in[13];
  const float* out_b     = (const float*)d_in[14];
  float* out = (float*)d_out;

  char* ws = (char*)d_ws;
  const size_t MB = 1ull << 20;
  _Float16* xn16  = (_Float16*)(ws + 0);          // 16 MB  (reused by `gated` later)
  _Float16* x16   = (_Float16*)(ws + 16 * MB);    // 32 MB
  _Float16* z16   = (_Float16*)(ws + 48 * MB);    // 32 MB
  _Float16* xc16  = (_Float16*)(ws + 80 * MB);    // 32 MB
  float*    xg    = (float*)   (ws + 112 * MB);   //  3 MB
  _Float16* ypad  = (_Float16*)(ws + 115 * MB);   //  1 MB
  _Float16* w1_16 = (_Float16*)(ws + 116 * MB);   //  2 MB
  _Float16* wg16  = (_Float16*)(ws + 118 * MB);   // 96 KB
  _Float16* wsm16 = (_Float16*)(ws + 119 * MB);   // 64 KB
  _Float16* wo16  = (_Float16*)(ws + 120 * MB);   //  1 MB
  _Float16* gated = (_Float16*)(ws + 0);          // 32 MB, reuses xn16/x16 region

  // weight conversions (cheap, deterministic per call)
  k_f32_to_f16<<<(2 * DI * DM + 255) / 256, 256, 0, stream>>>(in_proj_w, w1_16, 2 * DI * DM);
  k_f32_to_f16<<<(48 * DI + 255) / 256, 256, 0, stream>>>(gru_w_ih, wg16, 48 * DI);
  k_f32_to_f16<<<(DM * DI + 255) / 256, 256, 0, stream>>>(out_w, wo16, DM * DI);
  k_ssm_pad<<<(DI * 32 + 255) / 256, 256, 0, stream>>>(ssm_w, wsm16);

  // 1. layernorm -> f16
  k_layernorm<<<MROWS, 256, 0, stream>>>(x, norm_w, norm_b, xn16);
  // 2. in_proj GEMM (M=16384,N=2048,K=512), split x_/z : 512*32 waves / 8 per block
  k_gemm_inproj<<<(MT2 * 32) / 8, 256, 0, stream>>>(xn16, w1_16, in_proj_b, x16, z16);
  // 3. depthwise conv + SiLU
  k_conv_silu<<<(MROWS * DI) / 256, 256, 0, stream>>>(x16, conv_w, conv_b, xc16);
  // 4. GRU input GEMM (M=16384,N=48,K=1024)
  k_gemm_gru<<<MT2 / 8, 256, 0, stream>>>(xc16, wg16, gru_b_ih, xg);
  // 5. serial GRU scan
  k_gru_scan<<<1, 128, 0, stream>>>(xg, gru_w_hh, gru_b_hh, ypad);
  // 6. ssm_proj GEMM (M=16384,N=1024,K=32) fused with silu(z) gating
  k_gemm_ssm_gate<<<(MT2 * 16) / 8, 256, 0, stream>>>(ypad, wsm16, ssm_b, z16, gated);
  // 7. out_proj GEMM (M=16384,N=512,K=1024) + bias + residual
  k_gemm_out<<<(MT2 * 8) / 8, 256, 0, stream>>>(gated, wo16, out_b, x, out);
}
